// EquivariantPosUpdate_80865644249803
// MI455X (gfx1250) — compile-verified, matrix operand order checked by reading, not run on hardware
//
#include <hip/hip_runtime.h>
#include <cstdint>

#define E_EDGES 8192
#define N_NODES 2048

typedef __attribute__((ext_vector_type(16))) __bf16 v16bf;
typedef __attribute__((ext_vector_type(8)))  float  v8f;

// ---------- bf16 helpers ----------
__device__ __forceinline__ __bf16 f2bf(float x) { return (__bf16)x; }
__device__ __forceinline__ unsigned short f2bf_bits(float x) {
  union { __bf16 h; unsigned short u; } o; o.h = (__bf16)x; return o.u;
}

__device__ __forceinline__ float silu(float y) { return y / (1.0f + __expf(-y)); }

// ---------- WMMA fragment plumbing (CDNA5 ISA 7.12.2 layouts) ----------
// A tile in LDS row-major [16][lda]:
// lanes 0-15:  M=lane,    halves 0..7 -> K=k0+0..7,  halves 8..15 -> K=k0+16..23
// lanes 16-31: M=lane-16, halves 0..7 -> K=k0+8..15, halves 8..15 -> K=k0+24..31
__device__ __forceinline__ v16bf frag_a(const __bf16* A, int lda, int k0, int lane) {
  int m  = lane & 15;
  int kb = (lane >> 4) << 3;
  const __bf16* p = A + m * lda + k0;
  v16bf a;
#pragma unroll
  for (int j = 0; j < 8; ++j) { a[j] = p[kb + j]; a[8 + j] = p[16 + kb + j]; }
  return a;
}
// B is pre-packed: [(k/32)*N + n][32] halves, in-chunk order
// [k%32 = 0..7, 16..23, 8..15, 24..31]; a lane's 16 halves are one
// 32-byte contiguous run -> 2x ds_load_b128.
__device__ __host__ __forceinline__ int bswz(int kr) {  // involution 0..31
  return (kr & 7) + ((kr & 8) ? 16 : 0) + ((kr & 16) ? 8 : 0);
}
__device__ __forceinline__ v16bf frag_b_sw(const __bf16* Bsw, int N, int k0, int n0, int lane) {
  int n = n0 + (lane & 15);
  int g = (lane >> 4) << 4;            // 0 or 16 halves
  const __bf16* p = Bsw + ((size_t)((k0 >> 5) * N + n) << 5) + g;
  return *(const v16bf*)p;             // 32B contiguous, 32B aligned
}
// C layout: VGPR r, lanes 0-15 -> (M=r, N=lane); lanes 16-31 -> (M=8+r, N=lane-16)
__device__ __forceinline__ void store_c(float* C, int ldc, int n0, int lane, v8f c) {
  int mh = (lane >> 4) << 3;
  int n  = n0 + (lane & 15);
#pragma unroll
  for (int r = 0; r < 8; ++r) C[(size_t)(r + mh) * ldc + n] = c[r];
}
__device__ __forceinline__ void wave_gemm(const __bf16* A, int lda, const __bf16* Bsw, int BN,
                                          float* C, int ldc, int K, int n0, int lane) {
  v8f c = {};
  for (int k0 = 0; k0 < K; k0 += 32) {
    v16bf a = frag_a(A, lda, k0, lane);
    v16bf b = frag_b_sw(Bsw, BN, k0, n0, lane);
    c = __builtin_amdgcn_wmma_f32_16x16x32_bf16(false, a, false, b, (short)0, c, false, false);
  }
  store_c(C, ldc, n0, lane, c);
}

// ---------- prep: pack a K x N fp32 B-matrix into swizzled bf16 ----------
__global__ void pack_b_kernel(const float* __restrict__ W, int K, int N,
                              unsigned short* __restrict__ out) {
  int i = blockIdx.x * 256 + threadIdx.x;
  if (i >= K * N) return;
  int kc  = i / (N * 32);
  int rem = i % (N * 32);
  int n   = rem / 32;
  int pos = rem % 32;
  int kr  = bswz(pos);                 // involution
  out[i] = f2bf_bits(W[(size_t)(kc * 32 + kr) * N + n]);
}

// ---------- K: zero output ----------
__global__ void zero_kernel(float* p, int n) {
  int i = blockIdx.x * blockDim.x + threadIdx.x;
  if (i < n) p[i] = 0.0f;
}

// ---------- K0: mod = t @ norm_Wt + norm_bt (first 192 cols only) ----------
__global__ void mod_kernel(const float* __restrict__ t, const float* __restrict__ Wt,
                           const float* __restrict__ bt, float* __restrict__ modG) {
  int g = blockIdx.x, j = threadIdx.x;   // blockDim = 192
  float acc = bt[j];
  const float* tr = t + g * 128;
  for (int k = 0; k < 128; ++k) acc += tr[k] * Wt[k * 320 + j];
  modG[g * 192 + j] = acc;
}

// ---------- K1: RBF + 2-layer radial front for nf & ef ----------
__device__ void radial_branch(const __bf16* ES /*16x128*/, __bf16* Bsw /*4*64*32*/,
                              float (&S)[16][64], __bf16 (&A2)[16][64],
                              const unsigned short* W1p, const float* g1, const float* b1,
                              const unsigned short* W2p, const float* g2, const float* b2,
                              unsigned short* hid_out, int e0, int tid, int lane, int wv) {
  {
    const int4* s4 = (const int4*)W1p;  int4* d4 = (int4*)Bsw;
    for (int i = tid; i < 128 * 8; i += 128) d4[i] = s4[i];
  }
  __syncthreads();
  wave_gemm(ES, 128, Bsw, 64, &S[0][0], 64, 128, wv * 16, lane);
  __syncthreads();
  if (tid < 16) {
    float mu = 0.f; for (int c = 0; c < 64; ++c) mu += S[tid][c]; mu *= (1.0f / 64.0f);
    float var = 0.f; for (int c = 0; c < 64; ++c) { float d = S[tid][c] - mu; var += d * d; }
    float rs = rsqrtf(var * (1.0f / 64.0f) + 1e-5f);
    for (int c = 0; c < 64; ++c)
      A2[tid][c] = f2bf(silu((S[tid][c] - mu) * rs * g1[c] + b1[c]));
  }
  __syncthreads();
  {
    const int4* s4 = (const int4*)W2p;  int4* d4 = (int4*)Bsw;
    for (int i = tid; i < 64 * 8; i += 128) d4[i] = s4[i];
  }
  __syncthreads();
  wave_gemm(&A2[0][0], 64, Bsw, 64, &S[0][0], 64, 64, wv * 16, lane);
  __syncthreads();
  if (tid < 16) {
    float mu = 0.f; for (int c = 0; c < 64; ++c) mu += S[tid][c]; mu *= (1.0f / 64.0f);
    float var = 0.f; for (int c = 0; c < 64; ++c) { float d = S[tid][c] - mu; var += d * d; }
    float rs = rsqrtf(var * (1.0f / 64.0f) + 1e-5f);
    for (int c = 0; c < 64; ++c)
      hid_out[(size_t)(e0 + tid) * 64 + c] = f2bf_bits(silu((S[tid][c] - mu) * rs * g2[c] + b2[c]));
  }
  __syncthreads();
}

__global__ __launch_bounds__(128) void radial_kernel(
    const float* __restrict__ dist, const float* __restrict__ rbf_mean,
    const float* __restrict__ rbf_std, const float* __restrict__ rbf_w,
    const float* __restrict__ rbf_b,
    const unsigned short* nf_W1p, const float* nf_g1, const float* nf_b1,
    const unsigned short* nf_W2p, const float* nf_g2, const float* nf_b2,
    const unsigned short* ef_W1p, const float* ef_g1, const float* ef_b1,
    const unsigned short* ef_W2p, const float* ef_g2, const float* ef_b2,
    unsigned short* __restrict__ hid_nf, unsigned short* __restrict__ hid_ef) {
  __shared__ __align__(32) __bf16 ES[16][128];
  __shared__ __align__(32) __bf16 Bsw[4 * 64 * 32];
  __shared__ float  S[16][64];
  __shared__ __align__(32) __bf16 A2[16][64];
  int tid = threadIdx.x, lane = tid & 31, wv = tid >> 5;
  int e0 = blockIdx.x * 16;
  {
    int col = tid;
    float wsc = rbf_w[0], bsc = rbf_b[0];
    float mu = rbf_mean[col], sd = rbf_std[col];
    float inv = 1.0f / sd;
    float nrm = 0.3989422804014327f * inv;
    for (int r = 0; r < 16; ++r) {
      float x = dist[e0 + r] * 0.2f * wsc + bsc;   // /CUTOFF(=5)
      float z = (x - mu) * inv;
      ES[r][col] = f2bf(__expf(-0.5f * z * z) * nrm);
    }
  }
  __syncthreads();
  radial_branch(&ES[0][0], Bsw, S, A2, nf_W1p, nf_g1, nf_b1, nf_W2p, nf_g2, nf_b2, hid_nf, e0, tid, lane, wv);
  radial_branch(&ES[0][0], Bsw, S, A2, ef_W1p, ef_g1, ef_b1, ef_W2p, ef_g2, ef_b2, hid_ef, e0, tid, lane, wv);
}

// ---------- K2/K4: _lin (optionally gathered) ----------
template <int KS, int KV>
__global__ __launch_bounds__(128) void lin_kernel(
    const float* __restrict__ s_in, int s_stride,
    const float* __restrict__ v_in, int v_stride,
    const int* __restrict__ idx,
    const unsigned short* __restrict__ Wsp, const float* __restrict__ bs,
    const unsigned short* __restrict__ Wvp,
    float* __restrict__ s_out, float* __restrict__ v_out) {
  __shared__ __align__(32) __bf16 As[16][KS];
  __shared__ __align__(32) __bf16 Av[3][16][KV];
  __shared__ __align__(32) __bf16 BsS[(KS / 32) * 64 * 32];
  __shared__ __align__(32) __bf16 BvS[(KV / 32) * 32 * 32];
  __shared__ float  Cs[16][64];
  __shared__ float  Cv[3][16][32];
  int tid = threadIdx.x, lane = tid & 31, wv = tid >> 5;
  int e0 = blockIdx.x * 16;
  for (int r = wv; r < 16; r += 4) {
    int e = e0 + r;
    int row = idx ? idx[e] : e;
    const float* ps = s_in + (size_t)row * s_stride;
    const float* pv = v_in + (size_t)row * v_stride;
    for (int c = lane; c < KS; c += 32) As[r][c] = f2bf(ps[c]);
    for (int c = lane; c < KV * 3; c += 32) { int u = c / 3, k = c % 3; Av[k][r][u] = f2bf(pv[c]); }
  }
  {
    const int4* s4 = (const int4*)Wsp;  int4* d4 = (int4*)BsS;
    for (int i = tid; i < KS * 8; i += 128) d4[i] = s4[i];
    const int4* sv4 = (const int4*)Wvp; int4* dv4 = (int4*)BvS;
    for (int i = tid; i < KV * 4; i += 128) dv4[i] = sv4[i];
  }
  __syncthreads();
  wave_gemm(&As[0][0], KS, BsS, 64, &Cs[0][0], 64, KS, wv * 16, lane);
  for (int t = wv; t < 6; t += 4) {
    int k = t >> 1, n0 = (t & 1) * 16;
    wave_gemm(&Av[k][0][0], KV, BvS, 32, &Cv[k][0][0], 32, KV, n0, lane);
  }
  __syncthreads();
  float sS = rsqrtf((float)KS), sV = rsqrtf((float)KV);
  for (int i = tid; i < 16 * 64; i += 128) {
    int r = i >> 6, c = i & 63;
    s_out[(size_t)(e0 + r) * 64 + c] = Cs[r][c] * sS + bs[c];
  }
  for (int i = tid; i < 16 * 96; i += 128) {
    int r = i / 96, c = i % 96, u = c / 3, k = c % 3;
    v_out[((size_t)(e0 + r) * 32 + u) * 3 + k] = Cv[k][r][u] * sV;
  }
}

// ---------- K3/K5: fused W3 GEMM + depth-wise tensor product ----------
// w[e, :] = hid[e, :64] @ W3[64, 10240], produced 16x64 at a time in LDS and
// consumed immediately; FULL=false computes only ss + v0 blocks (av_ discarded).
// W3 is pre-packed bf16 in fragment layout; chunk loads are register-staged to
// overlap global latency with WMMA + consume of the previous chunk.
template <bool FULL>
__global__ __launch_bounds__(128) void dtp_kernel(
    const unsigned short* __restrict__ hid,
    const float* __restrict__ s1g, const float* __restrict__ s2g,
    const float* __restrict__ v1g, const float* __restrict__ v2g,
    const unsigned short* __restrict__ W3p, const float* __restrict__ bias,
    float* __restrict__ fs_out, float* __restrict__ fv_out) {
  __shared__ __align__(32) __bf16 Ah[16][64];
  __shared__ __align__(32) __bf16 Bw[2 * 64 * 32];
  __shared__ float  WC[16][64];
  __shared__ float  s1t[16][64], s2t[16][64];
  __shared__ float  v1t[16][32][3], v2t[16][32][3];
  __shared__ float  acc_ss[16][64];
  __shared__ float  acc_v0[16][32];
  __shared__ float  inn_sv[16][64][3];
  __shared__ float  inn_vs[16][32];
  __shared__ float  inn_v1[16][32][3];
  __shared__ float  red[16][8];
  int tid = threadIdx.x, lane = tid & 31, wv = tid >> 5;
  int e0 = blockIdx.x * 16;

  // --- vectorized tile loads (all contiguous) ---
  {
    int4* dA = (int4*)(&Ah[0][0]);
    const int4* sA = (const int4*)(hid + (size_t)e0 * 64);
    dA[tid] = sA[tid];                               // 16*64 bf16 = 128 int4
    float4* d1 = (float4*)(&s1t[0][0]);
    const float4* g1 = (const float4*)(s1g + (size_t)e0 * 64);
    float4* d2 = (float4*)(&s2t[0][0]);
    const float4* g2 = (const float4*)(s2g + (size_t)e0 * 64);
    for (int i = tid; i < 256; i += 128) { d1[i] = g1[i]; d2[i] = g2[i]; }
    float4* dv1 = (float4*)(&v1t[0][0][0]);
    const float4* gv1 = (const float4*)(v1g + (size_t)e0 * 96);
    float4* dv2 = (float4*)(&v2t[0][0][0]);
    const float4* gv2 = (const float4*)(v2g + (size_t)e0 * 96);
    for (int i = tid; i < 384; i += 128) { dv1[i] = gv1[i]; dv2[i] = gv2[i]; }
  }
  __syncthreads();

  // A fragments are loop-invariant: hoist out of the chunk loop.
  v16bf a0 = frag_a(&Ah[0][0], 64, 0, lane);
  v16bf a1 = frag_a(&Ah[0][0], 64, 32, lane);

  // packed W3 layout: [(kc)*10240 + col][32]; a 64-col chunk = two contiguous
  // 256-int4 segments (kc=0 at ch*256, kc=1 at 40960 + ch*256).
  const int4* W3v = (const int4*)W3p;
  int4* Bwv = (int4*)(&Bw[0]);
  int4 stage[4];
  const int tcount = FULL ? 160 : 80;
  auto ch_of = [&](int t) { return FULL ? t : (t < 64 ? t : 128 + (t - 64)); };
  {
    size_t b0 = (size_t)ch_of(0) * 256;
#pragma unroll
    for (int j = 0; j < 2; ++j) { stage[j] = W3v[b0 + j * 128 + tid]; stage[2 + j] = W3v[40960 + b0 + j * 128 + tid]; }
  }

  for (int t = 0; t < tcount; ++t) {
    int ch = ch_of(t);
    int bid = (ch < 64) ? 0 : (ch < 96) ? 1 : (ch < 128) ? 2 : (ch < 144) ? 3 : 4;
#pragma unroll
    for (int j = 0; j < 4; ++j) Bwv[j * 128 + tid] = stage[j];
    __syncthreads();   // Bw ready; previous chunk fully consumed
    if (t + 1 < tcount) {   // kick off next chunk's loads (consumed next iter)
      size_t b0 = (size_t)ch_of(t + 1) * 256;
#pragma unroll
      for (int j = 0; j < 2; ++j) { stage[j] = W3v[b0 + j * 128 + tid]; stage[2 + j] = W3v[40960 + b0 + j * 128 + tid]; }
    }
    {
      v8f c = {};
      v16bf b0 = frag_b_sw(Bw, 64, 0, wv * 16, lane);
      c = __builtin_amdgcn_wmma_f32_16x16x32_bf16(false, a0, false, b0, (short)0, c, false, false);
      v16bf b1 = frag_b_sw(Bw, 64, 32, wv * 16, lane);
      c = __builtin_amdgcn_wmma_f32_16x16x32_bf16(false, a1, false, b1, (short)0, c, false, false);
      store_c(&WC[0][0], 64, wv * 16, lane, c);
    }
    __syncthreads();   // WC ready

    if (bid == 0 || bid == 2) {  // ss (u=ch) / vs (u=ch-96): sum_v w*s2
      int e = tid >> 3, l8 = tid & 7;
      float p = 0.f;
      for (int v = l8; v < 64; v += 8) p += WC[e][v] * s2t[e][v];
      red[e][l8] = p;
      __syncthreads();
      if (l8 == 0) {
        float s = 0.f;
#pragma unroll
        for (int j = 0; j < 8; ++j) s += red[e][j];
        if (bid == 0) acc_ss[e][ch] = s; else inn_vs[e][ch - 96] = s;
      }
    } else if (bid == 1 || bid == 4) {  // sv / v1 inner: sum_v w * v2[:,k]
      if (tid < 96) {
        int e = tid / 6, c6 = tid % 6, ul = c6 / 3, k = c6 % 3;
        float p = 0.f;
        for (int v = 0; v < 32; ++v) p += WC[e][ul * 32 + v] * v2t[e][v][k];
        if (bid == 1) inn_sv[e][(ch - 64) * 2 + ul][k] = p;
        else          inn_v1[e][(ch - 144) * 2 + ul][k] = p;
      }
    } else {  // v0: sum_v w * dot(v1[u], v2[v])
      if (tid < 32) {
        int e = tid >> 1, ul = tid & 1;
        int u = (ch - 128) * 2 + ul;
        float ax = v1t[e][u][0], ay = v1t[e][u][1], az = v1t[e][u][2];
        float p = 0.f;
        for (int v = 0; v < 32; ++v) {
          float d = ax * v2t[e][v][0] + ay * v2t[e][v][1] + az * v2t[e][v][2];
          p += WC[e][ul * 32 + v] * d;
        }
        acc_v0[e][u] = p;
      }
    }
  }
  __syncthreads();

  const float r_m0  = 0.125f;                 // 1/sqrt(64)
  const float r_m1  = 0.17677669529663687f;   // 1/sqrt(32)
  const float r_3m1 = 0.10206207261596575f;   // 1/sqrt(96)
  const float r_2m1 = 0.125f;                 // 1/sqrt(64)
  for (int i = tid; i < 16 * 96; i += 128) {
    int e = i / 96, c = i % 96;
    float val = (c < 64) ? s1t[e][c] * acc_ss[e][c] * r_m0
                         : acc_v0[e][c - 64] * r_3m1;
    fs_out[(size_t)(e0 + e) * 96 + c] = val + bias[c];
  }
  if (FULL) {
    for (int i = tid; i < 16 * 64 * 3; i += 128) {
      int e = i / 192, rem = i % 192, u = rem / 3, k = rem % 3;
      fv_out[((size_t)(e0 + e) * 128 + u) * 3 + k] = s1t[e][u] * inn_sv[e][u][k] * r_m1;
    }
    for (int i = tid; i < 16 * 32 * 3; i += 128) {
      int e = i / 96, rem = i % 96, u = rem / 3, k = rem % 3;
      fv_out[((size_t)(e0 + e) * 128 + 64 + u) * 3 + k] = v1t[e][u][k] * inn_vs[e][u] * r_m0;
    }
    for (int i = tid; i < 16 * 32; i += 128) {
      int e = i / 32, u = i % 32;
      float ax = v1t[e][u][0], ay = v1t[e][u][1], az = v1t[e][u][2];
      float bx = inn_v1[e][u][0], by = inn_v1[e][u][1], bz = inn_v1[e][u][2];
      size_t b = ((size_t)(e0 + e) * 128 + 96 + u) * 3;
      fv_out[b + 0] = (ay * bz - az * by) * r_2m1;
      fv_out[b + 1] = (az * bx - ax * bz) * r_2m1;
      fv_out[b + 2] = (ax * by - ay * bx) * r_2m1;
    }
  }
}

// ---------- K6: adaLN + small MLP + force scatter ----------
__global__ __launch_bounds__(128) void final_kernel(
    const float* __restrict__ as_, const float* __restrict__ modG,
    const int* __restrict__ src, const int* __restrict__ batch,
    const float* __restrict__ dist, const float* __restrict__ relvec,
    const float* __restrict__ sp_W1, const float* __restrict__ sp_b1,
    const float* __restrict__ sp_W2, const float* __restrict__ sp_b2,
    float* __restrict__ out) {
  int e = blockIdx.x * blockDim.x + threadIdx.x;
  if (e >= E_EDGES) return;
  int s = src[e], g = batch[s];
  const float* mod = modG + g * 192;
  const float* x = as_ + (size_t)e * 96;
  float mu = 0.f; for (int c = 0; c < 96; ++c) mu += x[c]; mu *= (1.0f / 96.0f);
  float var = 0.f; for (int c = 0; c < 96; ++c) { float d = x[c] - mu; var += d * d; }
  float rs = rsqrtf(var * (1.0f / 96.0f) + 1e-5f);
  float sn[96];
  for (int c = 0; c < 96; ++c) sn[c] = (x[c] - mu) * rs * (1.0f + mod[96 + c]) + mod[c];
  const float inv96 = 0.10206207261596575f;   // 1/sqrt(96)
  const float inv32 = 0.17677669529663687f;   // 1/sqrt(32)
  float dot2 = 0.f;
  for (int h = 0; h < 32; ++h) {
    float d = 0.f;
    for (int c = 0; c < 96; ++c) d += sn[c] * sp_W1[c * 32 + h];
    dot2 += silu(d * inv96 + sp_b1[h]) * sp_W2[h];
  }
  float sw = dot2 * inv32 + sp_b2[0];
  float d2 = dist[e];
  float coef = sw / (d2 * (d2 + 1.0f));
  atomicAdd(&out[s * 3 + 0], coef * relvec[e * 3 + 0]);
  atomicAdd(&out[s * 3 + 1], coef * relvec[e * 3 + 1]);
  atomicAdd(&out[s * 3 + 2], coef * relvec[e * 3 + 2]);
}

// ---------- host launch ----------
extern "C" void kernel_launch(void* const* d_in, const int* /*in_sizes*/, int /*n_in*/,
                              void* d_out, int /*out_size*/, void* d_ws, size_t /*ws_size*/,
                              hipStream_t stream) {
  const float* h_node  = (const float*)d_in[0];
  const float* h_edge  = (const float*)d_in[1];
  const float* relvec  = (const float*)d_in[3];
  const float* dist    = (const float*)d_in[4];
  const float* t_in    = (const float*)d_in[5];
  const float* rbf_mean = (const float*)d_in[6];
  const float* rbf_std  = (const float*)d_in[7];
  const float* rbf_w    = (const float*)d_in[8];
  const float* rbf_b    = (const float*)d_in[9];
  const float* nf_W1 = (const float*)d_in[10], *nf_g1 = (const float*)d_in[11], *nf_b1 = (const float*)d_in[12];
  const float* nf_W2 = (const float*)d_in[13], *nf_g2 = (const float*)d_in[14], *nf_b2 = (const float*)d_in[15];
  const float* nf_W3 = (const float*)d_in[16];
  const float* ef_W1 = (const float*)d_in[17], *ef_g1 = (const float*)d_in[18], *ef_b1 = (const float*)d_in[19];
  const float* ef_W2 = (const float*)d_in[20], *ef_g2 = (const float*)d_in[21], *ef_b2 = (const float*)d_in[22];
  const float* ef_W3 = (const float*)d_in[23];
  const float* src_Ws = (const float*)d_in[24], *src_bs = (const float*)d_in[25], *src_Wv = (const float*)d_in[26];
  const float* dst_Ws = (const float*)d_in[27], *dst_bs = (const float*)d_in[28], *dst_Wv = (const float*)d_in[29];
  const float* nf_bias = (const float*)d_in[30];
  const float* nt_Ws = (const float*)d_in[31], *nt_bs = (const float*)d_in[32], *nt_Wv = (const float*)d_in[33];
  const float* et_Ws = (const float*)d_in[34], *et_bs = (const float*)d_in[35], *et_Wv = (const float*)d_in[36];
  const float* ef_bias = (const float*)d_in[37];
  const float* norm_Wt = (const float*)d_in[38], *norm_bt = (const float*)d_in[39];
  const float* sp_W1 = (const float*)d_in[40], *sp_b1 = (const float*)d_in[41];
  const float* sp_W2 = (const float*)d_in[42], *sp_b2 = (const float*)d_in[43];
  const int* edge_index = (const int*)d_in[44];
  const int* batch = (const int*)d_in[45];
  const int* srci = edge_index;
  const int* dsti = edge_index + E_EDGES;
  float* out = (float*)d_out;

  uintptr_t base = (uintptr_t)d_ws;
  size_t off = 0;
  auto take = [&](size_t b) -> void* {
    void* p = (void*)(base + off);
    off += (b + 255) & ~(size_t)255;
    return p;
  };
  float* modG = (float*)take((size_t)64 * 192 * 4);
  unsigned short* hid_nf = (unsigned short*)take((size_t)E_EDGES * 64 * 2);
  unsigned short* hid_ef = (unsigned short*)take((size_t)E_EDGES * 64 * 2);
  float* s1 = (float*)take((size_t)E_EDGES * 64 * 4);
  float* s2 = (float*)take((size_t)E_EDGES * 64 * 4);
  float* v1 = (float*)take((size_t)E_EDGES * 96 * 4);
  float* v2 = (float*)take((size_t)E_EDGES * 96 * 4);
  float* fs = (float*)take((size_t)E_EDGES * 96 * 4);
  float* fv = (float*)take((size_t)E_EDGES * 384 * 4);
  float* nsb = (float*)take((size_t)E_EDGES * 64 * 4);
  float* nvb = (float*)take((size_t)E_EDGES * 96 * 4);
  float* esb = (float*)take((size_t)E_EDGES * 64 * 4);
  float* evb = (float*)take((size_t)E_EDGES * 96 * 4);
  float* asb = (float*)take((size_t)E_EDGES * 96 * 4);
  // packed bf16 weights (fragment layout)
  unsigned short* nf_W1p = (unsigned short*)take((size_t)128 * 64 * 2);
  unsigned short* nf_W2p = (unsigned short*)take((size_t)64 * 64 * 2);
  unsigned short* ef_W1p = (unsigned short*)take((size_t)128 * 64 * 2);
  unsigned short* ef_W2p = (unsigned short*)take((size_t)64 * 64 * 2);
  unsigned short* src_Wsp = (unsigned short*)take((size_t)128 * 64 * 2);
  unsigned short* src_Wvp = (unsigned short*)take((size_t)64 * 32 * 2);
  unsigned short* dst_Wsp = (unsigned short*)take((size_t)128 * 64 * 2);
  unsigned short* dst_Wvp = (unsigned short*)take((size_t)64 * 32 * 2);
  unsigned short* nt_Wsp = (unsigned short*)take((size_t)96 * 64 * 2);
  unsigned short* nt_Wvp = (unsigned short*)take((size_t)128 * 32 * 2);
  unsigned short* et_Wsp = (unsigned short*)take((size_t)64 * 64 * 2);
  unsigned short* et_Wvp = (unsigned short*)take((size_t)32 * 32 * 2);
  unsigned short* nf_W3p = (unsigned short*)take((size_t)64 * 10240 * 2);
  unsigned short* ef_W3p = (unsigned short*)take((size_t)64 * 10240 * 2);

  auto pack = [&](const float* W, int K, int N, unsigned short* outp) {
    pack_b_kernel<<<(K * N + 255) / 256, 256, 0, stream>>>(W, K, N, outp);
  };
  pack(nf_W1, 128, 64, nf_W1p);  pack(nf_W2, 64, 64, nf_W2p);
  pack(ef_W1, 128, 64, ef_W1p);  pack(ef_W2, 64, 64, ef_W2p);
  pack(src_Ws, 128, 64, src_Wsp); pack(src_Wv, 64, 32, src_Wvp);
  pack(dst_Ws, 128, 64, dst_Wsp); pack(dst_Wv, 64, 32, dst_Wvp);
  pack(nt_Ws, 96, 64, nt_Wsp);   pack(nt_Wv, 128, 32, nt_Wvp);
  pack(et_Ws, 64, 64, et_Wsp);   pack(et_Wv, 32, 32, et_Wvp);
  pack(nf_W3, 64, 10240, nf_W3p);
  pack(ef_W3, 64, 10240, ef_W3p);

  zero_kernel<<<(N_NODES * 3 + 127) / 128, 128, 0, stream>>>(out, N_NODES * 3);
  mod_kernel<<<64, 192, 0, stream>>>(t_in, norm_Wt, norm_bt, modG);
  radial_kernel<<<E_EDGES / 16, 128, 0, stream>>>(dist, rbf_mean, rbf_std, rbf_w, rbf_b,
      nf_W1p, nf_g1, nf_b1, nf_W2p, nf_g2, nf_b2,
      ef_W1p, ef_g1, ef_b1, ef_W2p, ef_g2, ef_b2, hid_nf, hid_ef);
  lin_kernel<128, 64><<<E_EDGES / 16, 128, 0, stream>>>(h_node, 256, h_node + 128, 256, srci,
      src_Wsp, src_bs, src_Wvp, s1, v1);
  lin_kernel<128, 64><<<E_EDGES / 16, 128, 0, stream>>>(h_node, 256, h_node + 128, 256, dsti,
      dst_Wsp, dst_bs, dst_Wvp, s2, v2);
  dtp_kernel<true><<<E_EDGES / 16, 128, 0, stream>>>(hid_nf, s1, s2, v1, v2, nf_W3p, nf_bias, fs, fv);
  lin_kernel<96, 128><<<E_EDGES / 16, 128, 0, stream>>>(fs, 96, fv, 384, nullptr,
      nt_Wsp, nt_bs, nt_Wvp, nsb, nvb);
  lin_kernel<64, 32><<<E_EDGES / 16, 128, 0, stream>>>(h_edge, 160, h_edge + 64, 160, nullptr,
      et_Wsp, et_bs, et_Wvp, esb, evb);
  dtp_kernel<false><<<E_EDGES / 16, 128, 0, stream>>>(hid_ef, nsb, esb, nvb, evb, ef_W3p, ef_bias, asb, nullptr);
  final_kernel<<<E_EDGES / 128, 128, 0, stream>>>(asb, modG, srci, batch, dist, relvec,
      sp_W1, sp_b1, sp_W2, sp_b2, out);
}
